// CRF_4922032521709
// MI455X (gfx1250) — compile-verified, matrix operand order checked by reading, not run on hardware
//
#include <hip/hip_runtime.h>

// CRF forward scan in exp-space via WMMA.
// Layout invariants (CDNA5 wave32, 16x16x32 f16 WMMA):
//   A (16xK f16):  lane L holds row m = L&15; element e -> K = 16*(e>>3) + 8*(L>>4) + (e&7)
//   B (Kx16 f16):  lane L holds col n = L&15; element e -> K = 16*(e>>3) + 8*(L>>4) + (e&7)
//   C/D (16x16 f32): lane L holds col n = L&15; VGPR r -> row m = r + 8*(L>>4)
// We compute D[j, b] = sum_k exp(trans[j,k]-Tmax) * exp(alpha[b,k]-bmax_b)
// so D's layout (lane = batch col, r = tag row) equals the B-operand layout of
// the next step's exp(alpha): the recurrent state needs NO cross-lane movement.

typedef __attribute__((ext_vector_type(16))) _Float16 v16h;
typedef __attribute__((ext_vector_type(8)))  float    v8f;

#define T_TAGS 64

__global__ __launch_bounds__(32) void crf_fwd_wmma(
    const float* __restrict__ feats,   // [B, S, T]
    const float* __restrict__ masks,   // [B, S]
    const float* __restrict__ trans,   // [T, T]
    float* __restrict__ out,           // [B, T]
    int S)
{
  const float L2E = 1.4426950408889634f;   // log2(e)
  const float LN2 = 0.6931471805599453f;   // ln(2)

  const int lane = threadIdx.x & 31;
  const int bcol = lane & 15;          // batch column within this 16-row tile
  const int half = lane >> 4;          // 0/1: selects tag sub-rows r+8*half
  const int b    = blockIdx.x * 16 + bcol;

  // ---- global max of transitions (for exp-space scaling) ----
  float tmax = -3.402823e38f;
  for (int i = lane; i < T_TAGS * T_TAGS; i += 32)
    tmax = fmaxf(tmax, trans[i]);
#pragma unroll
  for (int m = 16; m >= 1; m >>= 1)
    tmax = fmaxf(tmax, __shfl_xor(tmax, m, 32));

  // ---- loop-invariant A operands: exp(trans - tmax), WMMA A layout ----
  // Tile u covers out-tags j = 16u + bcol; chunk c covers in-tags 32c..32c+31.
  v16h eT[4][2];
#pragma unroll
  for (int u = 0; u < 4; ++u)
#pragma unroll
    for (int c = 0; c < 2; ++c)
#pragma unroll
      for (int e = 0; e < 16; ++e) {
        const int j = 16 * u + bcol;
        const int k = 32 * c + 16 * (e >> 3) + 8 * half + (e & 7);
        eT[u][c][e] =
            (_Float16)__builtin_amdgcn_exp2f((trans[j * T_TAGS + k] - tmax) * L2E);
      }

  // alpha[t][r] = alpha[b][tag] with tag = r + 8*half + 16*t  (D-matrix layout)
  float alpha[4][8];
  float fA[4][8], fB[4][8];

  const float* fbase = feats + (size_t)b * S * T_TAGS + 8 * half;

  auto loadFeat = [&](float (&f)[4][8], int s) {
    const float* p = fbase + (size_t)s * T_TAGS;
#pragma unroll
    for (int t = 0; t < 4; ++t) {
      float4 lo = *(const float4*)(p + 16 * t);
      float4 hi = *(const float4*)(p + 16 * t + 4);
      f[t][0] = lo.x; f[t][1] = lo.y; f[t][2] = lo.z; f[t][3] = lo.w;
      f[t][4] = hi.x; f[t][5] = hi.y; f[t][6] = hi.z; f[t][7] = hi.w;
    }
  };

  loadFeat(alpha, 0);   // alpha0 = feats[:, 0]
  loadFeat(fA, 1);      // prefetch step 1 features

  auto stepBody = [&](float (&fc)[4][8], float (&fn)[4][8], int s) {
    if (s + 1 < S) loadFeat(fn, s + 1);          // hide HBM latency of next step
    const float mval = masks[(size_t)b * S + s];
    const float oneMinusM = 1.0f - mval;

    // per-batch-column max over all 64 tags (two lanes share one batch col)
    float bm = -3.402823e38f;
#pragma unroll
    for (int t = 0; t < 4; ++t)
#pragma unroll
      for (int r = 0; r < 8; ++r) bm = fmaxf(bm, alpha[t][r]);
    bm = fmaxf(bm, __shfl_xor(bm, 16, 32));

    // exp(alpha - bm) = exp2(alpha*log2e - bm*log2e): 1 fma + 1 v_exp each
    const float nbm = -bm * L2E;
    v16h B0, B1;
#pragma unroll
    for (int e = 0; e < 16; ++e) {
      B0[e] = (_Float16)__builtin_amdgcn_exp2f(
                  fmaf(alpha[(e >> 3)][e & 7], L2E, nbm));
      B1[e] = (_Float16)__builtin_amdgcn_exp2f(
                  fmaf(alpha[2 + (e >> 3)][e & 7], L2E, nbm));
    }

    const float base  = bm + tmax;        // added back after log
    const float fallb = bm - 10000.0f;    // underflow fallback (~= reference)

    const v8f zero = {};
#pragma unroll
    for (int u = 0; u < 4; ++u) {
      v8f acc = __builtin_amdgcn_wmma_f32_16x16x32_f16(
          false, eT[u][0], false, B0, (short)0, zero, false, false);
      acc = __builtin_amdgcn_wmma_f32_16x16x32_f16(
          false, eT[u][1], false, B1, (short)0, acc, false, false);
#pragma unroll
      for (int r = 0; r < 8; ++r) {
        float d   = acc[r];
        // branchless: raw v_log_f32 (log2) * ln2 + base, cndmask on underflow
        float lse = fmaf(__builtin_amdgcn_logf(d), LN2, base);
        lse = (d > 0.0f) ? lse : fallb;
        float na = lse + fc[u][r];
        alpha[u][r] = na * mval + alpha[u][r] * oneMinusM;
      }
    }
  };

  int s = 1;
  for (; s + 1 < S; s += 2) {       // unrolled x2 to ping-pong feat buffers
    stepBody(fA, fB, s);
    stepBody(fB, fA, s + 1);
  }
  if (s < S) stepBody(fA, fB, s);

  // write final alpha: out[b][tag], tag = r + 8*half + 16*t
  float* op = out + (size_t)b * T_TAGS + 8 * half;
#pragma unroll
  for (int t = 0; t < 4; ++t) {
    float4 lo = {alpha[t][0], alpha[t][1], alpha[t][2], alpha[t][3]};
    float4 hi = {alpha[t][4], alpha[t][5], alpha[t][6], alpha[t][7]};
    *(float4*)(op + 16 * t)     = lo;
    *(float4*)(op + 16 * t + 4) = hi;
  }
}

extern "C" void kernel_launch(void* const* d_in, const int* in_sizes, int n_in,
                              void* d_out, int out_size, void* d_ws, size_t ws_size,
                              hipStream_t stream) {
  (void)n_in; (void)out_size; (void)d_ws; (void)ws_size;
  const float* feats = (const float*)d_in[0];   // [B, S, T] f32
  const float* masks = (const float*)d_in[1];   // [B, S]    f32
  const float* trans = (const float*)d_in[2];   // [T, T]    f32

  const int S = 1024;                 // per reference setup
  const int B = in_sizes[1] / S;      // masks is B*S elements
  const int blocks = B / 16;          // one 16-batch tile per single-wave WG

  crf_fwd_wmma<<<blocks, 32, 0, stream>>>(feats, masks, trans,
                                          (float*)d_out, S);
}
